// Decoder_69260642615503
// MI455X (gfx1250) — matrix operand order repaired and checked by reading, not compiled, not run on hardware
//
#include <hip/hip_runtime.h>
#include <hip/hip_bf16.h>
#include <math.h>

#define V_   32000
#define Dm   512
#define Hm   512
#define Bm   32
#define Tm   128
#define ML   48
#define KX   1024   // D + H
#define G3   1536   // 3*H
#define NTILES (V_/16)   // 2000

typedef float  v2f  __attribute__((ext_vector_type(2)));
typedef float  v8f  __attribute__((ext_vector_type(8)));
typedef __bf16 bf16x16 __attribute__((ext_vector_type(16)));

// ---------------------------------------------------------------------------
// Generic fp32 WMMA GEMM: C[M][N] = A[M][K] @ W[N][K]^T (+ bias[n])
// A row-major stride K, W row-major stride K, C row-major stride N.
// One wave computes a 16x16 tile via v_wmma_f32_16x16x4_f32.
// A layout (16x4 f32): lanes 0-15 hold row m0+lane, K={k,k+1}; lanes 16-31 K={k+2,k+3}
// B layout (4x16 f32): lanes 0-15 hold col n0+lane, K={k,k+1}; lanes 16-31 K={k+2,k+3}
// ---------------------------------------------------------------------------
__global__ void wmma_gemm_f32(const float* __restrict__ A,
                              const float* __restrict__ W,
                              const float* __restrict__ bias,
                              float* __restrict__ C,
                              int M, int N, int K) {
  int wave  = threadIdx.x >> 5;
  int lane  = threadIdx.x & 31;
  int ntile = blockIdx.x * (blockDim.x >> 5) + wave;
  int mtile = blockIdx.y;
  if (ntile * 16 >= N) return;            // wave-uniform
  int m0 = mtile * 16, n0 = ntile * 16;
  int lrow = lane & 15;
  int koff = (lane >= 16) ? 2 : 0;
  const float* arow = A + (size_t)(m0 + lrow) * K + koff;
  const float* wrow = W + (size_t)(n0 + lrow) * K + koff;
  v8f acc = {};
  for (int k = 0; k < K; k += 4) {
    v2f a = *(const v2f*)(arow + k);
    v2f b = *(const v2f*)(wrow + k);
    acc = __builtin_amdgcn_wmma_f32_16x16x4_f32(false, a, false, b,
                                                (short)0, acc, false, false);
  }
  float bv = bias ? bias[n0 + lrow] : 0.0f;
  int row0 = m0 + ((lane >= 16) ? 8 : 0);
  int col  = n0 + lrow;
  #pragma unroll
  for (int g = 0; g < 8; ++g)
    C[(size_t)(row0 + g) * N + col] = acc[g] + bv;
}

// ---------------------------------------------------------------------------
// One-time: repack lin_W (V x 1024 f32) -> bf16 in WMMA B-matrix tile order.
// Bpack unit = bf16x16 (32B per lane). Index: (tile*32 + kt)*32 + lane.
// lane<16 : col=tile*16+lane,    K = kt*32 + 0..15
// lane>=16: col=tile*16+lane-16, K = kt*32 + 16..31
// ---------------------------------------------------------------------------
__global__ void repack_linW(const float* __restrict__ linW, __bf16* __restrict__ Bpack) {
  int idx = blockIdx.x * blockDim.x + threadIdx.x;     // 2000*32*32
  if (idx >= NTILES * 32 * 32) return;
  int lane = idx & 31, kt = (idx >> 5) & 31, tile = idx >> 10;
  int v  = tile * 16 + (lane & 15);
  int kb = kt * 32 + ((lane < 16) ? 0 : 16);
  const float* src = linW + (size_t)v * KX + kb;
  __bf16* dst = Bpack + (size_t)idx * 16;
  #pragma unroll
  for (int i = 0; i < 16; ++i) dst[i] = (__bf16)src[i];
}

// ---------------------------------------------------------------------------
// Per step: pack A = [h_new | ctx] (32 x 1024 f32) -> bf16 WMMA A-layout.
// Apack unit bf16x16, index (mtile*32 + kt)*32 + lane.
// lane<16 : row=mtile*16+lane,    K = kt*32 + 0..15
// lane>=16: row=mtile*16+lane-16, K = kt*32 + 16..31
// ---------------------------------------------------------------------------
__global__ void pack_A(const float* __restrict__ h,
                       const float* __restrict__ xcat1,   // cols 512..1023 hold ctx
                       __bf16* __restrict__ Apack) {
  int idx = blockIdx.x * blockDim.x + threadIdx.x;     // 2*32*32 = 2048
  if (idx >= 2048) return;
  int lane = idx & 31, kt = (idx >> 5) & 31, mtile = idx >> 10;
  int r  = mtile * 16 + (lane & 15);
  int kb = kt * 32 + ((lane < 16) ? 0 : 16);
  __bf16* dst = Apack + (size_t)idx * 16;
  #pragma unroll
  for (int i = 0; i < 16; ++i) {
    int k = kb + i;                                    // kb..kb+15
    float x = (k < Hm) ? h[(size_t)r * Hm + k] : xcat1[(size_t)r * KX + k];
    dst[i] = (__bf16)x;
  }
}

// ---------------------------------------------------------------------------
// Logits GEMM: score[b][v] = [h,ctx]_bf16 @ lin_W_bf16^T + lin_b, b=0..31.
// One wave: one N-tile (16 vocab cols) x both M-tiles (reuses B operand).
// 32 K-iterations of v_wmma_f32_16x16x32_bf16; writes straight into d_out.
// ---------------------------------------------------------------------------
__global__ void logits_wmma(const __bf16* __restrict__ Apack,
                            const __bf16* __restrict__ Bpack,
                            const float* __restrict__ lin_b,
                            float* __restrict__ out, int step) {
  int wave = threadIdx.x >> 5, lane = threadIdx.x & 31;
  int ntile = blockIdx.x * (blockDim.x >> 5) + wave;
  if (ntile >= NTILES) return;                         // wave-uniform
  const bf16x16* bp = (const bf16x16*)Bpack + (size_t)ntile * 32 * 32 + lane;
  const bf16x16* ap = (const bf16x16*)Apack + lane;
  v8f c0 = {}, c1 = {};
  for (int kt = 0; kt < 32; ++kt) {
    bf16x16 b  = bp[kt * 32];
    bf16x16 a0 = ap[kt * 32];          // batch rows 0..15
    bf16x16 a1 = ap[(32 + kt) * 32];   // batch rows 16..31
    c0 = __builtin_amdgcn_wmma_f32_16x16x32_bf16(false, a0, false, b,
                                                 (short)0, c0, false, false);
    c1 = __builtin_amdgcn_wmma_f32_16x16x32_bf16(false, a1, false, b,
                                                 (short)0, c1, false, false);
  }
  int lrow = lane & 15;
  int v = ntile * 16 + lrow;
  float bv = lin_b[v];
  int b0 = (lane >= 16) ? 8 : 0;
  #pragma unroll
  for (int g = 0; g < 8; ++g) {
    int bb = b0 + g;
    out[((size_t)bb        * ML + step) * V_ + v] = c0[g] + bv;
    out[((size_t)(bb + 16) * ML + step) * V_ + v] = c1[g] + bv;
  }
}

// ---------------------------------------------------------------------------
// GRU gate fusion: h = (1-z)*n + z*h
// ---------------------------------------------------------------------------
__global__ void gru_gates(const float* __restrict__ gi, const float* __restrict__ gh,
                          const float* __restrict__ bih, const float* __restrict__ bhh,
                          float* __restrict__ h) {
  int idx = blockIdx.x * blockDim.x + threadIdx.x;     // 32*512
  if (idx >= Bm * Hm) return;
  int b = idx >> 9, j = idx & 511;
  const float* gib = gi + (size_t)b * G3;
  const float* ghb = gh + (size_t)b * G3;
  float ir = gib[j]           + bih[j];
  float iz = gib[Hm + j]      + bih[Hm + j];
  float in_= gib[2 * Hm + j]  + bih[2 * Hm + j];
  float hr = ghb[j]           + bhh[j];
  float hz = ghb[Hm + j]      + bhh[Hm + j];
  float hn = ghb[2 * Hm + j]  + bhh[2 * Hm + j];
  float r = 1.0f / (1.0f + expf(-(ir + hr)));
  float z = 1.0f / (1.0f + expf(-(iz + hz)));
  float n = tanhf(in_ + r * hn);
  float ho = h[idx];
  h[idx] = (1.0f - z) * n + z * ho;
}

// ---------------------------------------------------------------------------
// In-place log_softmax over a vocab row + argmax token. 1 block per batch row.
// ---------------------------------------------------------------------------
__global__ void softmax_argmax(float* __restrict__ out, int* __restrict__ tokens, int step) {
  int b = blockIdx.x, tid = threadIdx.x;               // 1024 threads
  float* row = out + ((size_t)b * ML + step) * V_;
  __shared__ float sred[1024];
  __shared__ int   sidx[1024];
  float mx = -3.402823466e38f; int mi = 0;
  for (int v = tid; v < V_; v += 1024) {
    float x = row[v];
    if (x > mx) { mx = x; mi = v; }
  }
  sred[tid] = mx; sidx[tid] = mi;
  __syncthreads();
  for (int s = 512; s > 0; s >>= 1) {
    if (tid < s) {
      float xo = sred[tid + s];
      if (xo > sred[tid] || (xo == sred[tid] && sidx[tid + s] < sidx[tid])) {
        sred[tid] = xo; sidx[tid] = sidx[tid + s];
      }
    }
    __syncthreads();
  }
  float rowmax = sred[0];
  if (tid == 0) tokens[b] = sidx[0];
  __syncthreads();
  float acc = 0.0f;
  for (int v = tid; v < V_; v += 1024) acc += expf(row[v] - rowmax);
  sred[tid] = acc;
  __syncthreads();
  for (int s = 512; s > 0; s >>= 1) {
    if (tid < s) sred[tid] += sred[tid + s];
    __syncthreads();
  }
  float lse = rowmax + logf(sred[0]);
  for (int v = tid; v < V_; v += 1024) row[v] = row[v] - lse;
}

// ---------------------------------------------------------------------------
// e_next = emb[argmax] into xcat1[:, 0:512]
// ---------------------------------------------------------------------------
__global__ void gather_e(const float* __restrict__ emb, const int* __restrict__ tokens,
                         float* __restrict__ xcat1) {
  int b = blockIdx.x, tid = threadIdx.x;               // 128 threads
  int tok = tokens[b];
  for (int j = tid; j < Dm; j += 128)
    xcat1[(size_t)b * KX + j] = emb[(size_t)tok * Dm + j];
}

// ---------------------------------------------------------------------------
// alpha = softmax_t(energies[b,t,:] . h[b,:]); ctx = alpha @ enc -> xcat1[:,512:]
// ---------------------------------------------------------------------------
__global__ void attention(const float* __restrict__ energies, const float* __restrict__ enc,
                          const float* __restrict__ h, float* __restrict__ xcat1) {
  int b = blockIdx.x, tid = threadIdx.x;               // 512 threads
  __shared__ float sc[Tm];
  __shared__ float tmp[Tm];
  if (tid < Tm) {
    const float* erow = energies + ((size_t)b * Tm + tid) * Hm;
    const float* hb = h + (size_t)b * Hm;
    float s = 0.0f;
    for (int k = 0; k < Hm; ++k) s += erow[k] * hb[k];
    sc[tid] = s; tmp[tid] = s;
  }
  __syncthreads();
  for (int s = Tm / 2; s > 0; s >>= 1) {
    if (tid < s) tmp[tid] = fmaxf(tmp[tid], tmp[tid + s]);
    __syncthreads();
  }
  float mx = tmp[0];
  __syncthreads();
  if (tid < Tm) { float e = expf(sc[tid] - mx); sc[tid] = e; tmp[tid] = e; }
  __syncthreads();
  for (int s = Tm / 2; s > 0; s >>= 1) {
    if (tid < s) tmp[tid] += tmp[tid + s];
    __syncthreads();
  }
  float inv = 1.0f / tmp[0];
  __syncthreads();
  float ctx = 0.0f;
  for (int t = 0; t < Tm; ++t)
    ctx += sc[t] * inv * enc[((size_t)b * Tm + t) * Hm + tid];
  xcat1[(size_t)b * KX + Hm + tid] = ctx;
}

// ---------------------------------------------------------------------------
// Init per call: xcat1 = [emb[inputs] | context], h = 0
// ---------------------------------------------------------------------------
__global__ void init_state(const int* __restrict__ inputs, const float* __restrict__ context,
                           const float* __restrict__ emb, float* __restrict__ xcat1,
                           float* __restrict__ h) {
  int idx = blockIdx.x * blockDim.x + threadIdx.x;     // 32768
  if (idx < Bm * KX) {
    int b = idx >> 10, k = idx & 1023;
    float v;
    if (k < Dm) v = emb[(size_t)inputs[b] * Dm + k];
    else        v = context[(size_t)b * Hm + (k - Dm)];
    xcat1[idx] = v;
  }
  if (idx < Bm * Hm) h[idx] = 0.0f;
}

extern "C" void kernel_launch(void* const* d_in, const int* in_sizes, int n_in,
                              void* d_out, int out_size, void* d_ws, size_t ws_size,
                              hipStream_t stream) {
  const int*   inputs  = (const int*)  d_in[0];
  const float* context = (const float*)d_in[1];
  // d_in[2] = max_len scalar (== 48, compiled in)
  const float* enc     = (const float*)d_in[3];
  const float* emb     = (const float*)d_in[4];
  const float* Wih     = (const float*)d_in[5];
  const float* Whh     = (const float*)d_in[6];
  const float* bih     = (const float*)d_in[7];
  const float* bhh     = (const float*)d_in[8];
  const float* linW    = (const float*)d_in[9];
  const float* linB    = (const float*)d_in[10];
  const float* attnW   = (const float*)d_in[11];
  const float* attnB   = (const float*)d_in[12];
  float* out = (float*)d_out;

  char* w = (char*)d_ws;
  __bf16* Bpack    = (__bf16*)w;  w += (size_t)NTILES * 32 * 32 * 32;  // 65,536,000 B
  float* energies  = (float*)w;   w += (size_t)Bm * Tm * Hm * 4;       // 8 MB
  float* gi        = (float*)w;   w += (size_t)Bm * G3 * 4;
  float* gh        = (float*)w;   w += (size_t)Bm * G3 * 4;
  float* h         = (float*)w;   w += (size_t)Bm * Hm * 4;
  float* xcat1     = (float*)w;   w += (size_t)Bm * KX * 4;
  __bf16* Apack    = (__bf16*)w;  w += (size_t)2 * 32 * 32 * 32;       // 64 KB
  int* tokens      = (int*)w;     w += 256;

  // One-time (per call) setup
  repack_linW<<<(NTILES * 32 * 32 + 255) / 256, 256, 0, stream>>>(linW, Bpack);
  init_state<<<(Bm * KX + 255) / 256, 256, 0, stream>>>(inputs, context, emb, xcat1, h);
  // energies: (B*T=4096) x 512 = enc @ attn_W^T + attn_b
  wmma_gemm_f32<<<dim3(8, 256), 128, 0, stream>>>(enc, attnW, attnB, energies,
                                                  Bm * Tm, Hm, Hm);

  for (int t = 0; t < ML; ++t) {
    // gi = [e,ctx] @ Wih^T ; gh = h @ Whh^T   (fp32 WMMA, bias folded in gates)
    wmma_gemm_f32<<<dim3(24, 2), 128, 0, stream>>>(xcat1, Wih, nullptr, gi, Bm, G3, KX);
    wmma_gemm_f32<<<dim3(24, 2), 128, 0, stream>>>(h,     Whh, nullptr, gh, Bm, G3, Hm);
    gru_gates<<<(Bm * Hm + 255) / 256, 256, 0, stream>>>(gi, gh, bih, bhh, h);
    pack_A<<<8, 256, 0, stream>>>(h, xcat1, Apack);
    logits_wmma<<<NTILES / 8, 256, 0, stream>>>(Apack, Bpack, linB, out, t);
    softmax_argmax<<<Bm, 1024, 0, stream>>>(out, tokens, t);
    gather_e<<<Bm, 128, 0, stream>>>(emb, tokens, xcat1);
    attention<<<Bm, 512, 0, stream>>>(energies, enc, h, xcat1);
  }
}
